// RPCM_56667798503470
// MI455X (gfx1250) — compile-verified, matrix operand order checked by reading, not run on hardware
//
#include <hip/hip_runtime.h>

// ---------------------------------------------------------------------------
// RPCM (scene-graph message passing), STEPS=1, on MI455X / gfx1250.
//
// Roofline: W_collect = 402 MB fp32 -> 17.3 us floor @ 23.3 TB/s.
// GEMM FLOPs = 129 G fp32. We run bf16 hi/lo-split WMMA (3 products) with
// f32 accumulation: near-fp32 precision at 16-bit-WMMA rate, weights read
// once as fp32 and split to bf16 during the LDS staging pass.
// v2: double-buffered LDS pipeline -- global prefetch for step k+1 issues
// before the WMMAs of step k; one workgroup barrier per K-step instead of 2.
// All attention maps are structural (1-I / one-hot / sparse mask) and are
// handled with closed forms + L2-resident gathers, never dense GEMMs.
// ---------------------------------------------------------------------------

typedef __attribute__((ext_vector_type(16))) __bf16 v16bf;
typedef __attribute__((ext_vector_type(8)))  __bf16 v8bf;
typedef __attribute__((ext_vector_type(8)))  float  v8f;

#define DIM    4096
#define N_OBJ  256
#define N_REL  1024
#define EPSF   1e-7f

#define BM 64
#define BN 64
#define BK 32
#define LDSTR 40   // 32 + 8 bf16 padding: 80B row stride, conflict-free 16B reads

union FragU { v16bf v; v8bf h[2]; };

// H[m][n] = relu( sum_k X[m][k] * W[n][k] + bias[n] )   (NT GEMM, K-major both)
__global__ __launch_bounds__(128)
void rpcm_gemm_relu(const float* __restrict__ X, const float* __restrict__ W,
                    const float* __restrict__ bias, float* __restrict__ H)
{
    __shared__ __bf16 sa_hi[2][BM * LDSTR];
    __shared__ __bf16 sa_lo[2][BM * LDSTR];
    __shared__ __bf16 sb_hi[2][BM * LDSTR];
    __shared__ __bf16 sb_lo[2][BM * LDSTR];

    const int tid  = threadIdx.x;
    const int wave = tid >> 5;      // 0..3, each wave owns 16 M-rows
    const int lane = tid & 31;
    const int l16  = lane & 15;
    const int hs   = lane >> 4;     // K half-select per documented 16-bit A layout

    const int m0 = blockIdx.y * BM;
    const int n0 = blockIdx.x * BN;

    v8f acc[4] = {};                // f32 accumulators, 16x16 each

    float4 ra[4], rb[4];            // global prefetch registers (one BK tile)

    auto load_tile = [&](int k0) {
        for (int p = 0; p < 4; ++p) {
            const int e   = p * 512 + tid * 4;   // 2048 elems / tile
            const int row = e >> 5;
            const int kc  = e & 31;
            ra[p] = *reinterpret_cast<const float4*>(&X[(size_t)(m0 + row) * DIM + k0 + kc]);
            rb[p] = *reinterpret_cast<const float4*>(&W[(size_t)(n0 + row) * DIM + k0 + kc]);
        }
    };
    auto split_store = [&](int buf) {
        for (int p = 0; p < 4; ++p) {
            const int e   = p * 512 + tid * 4;
            const int row = e >> 5;
            const int kc  = e & 31;
            const int o   = row * LDSTR + kc;
            const float ax[4] = {ra[p].x, ra[p].y, ra[p].z, ra[p].w};
            const float bx[4] = {rb[p].x, rb[p].y, rb[p].z, rb[p].w};
            for (int q = 0; q < 4; ++q) {
                const __bf16 ah = (__bf16)ax[q];
                const __bf16 bh = (__bf16)bx[q];
                sa_hi[buf][o + q] = ah;  sa_lo[buf][o + q] = (__bf16)(ax[q] - (float)ah);
                sb_hi[buf][o + q] = bh;  sb_lo[buf][o + q] = (__bf16)(bx[q] - (float)bh);
            }
        }
    };

    load_tile(0);
    split_store(0);
    __syncthreads();

    const int NK = DIM / BK;        // 128
    for (int k = 0; k < NK; ++k) {
        const int buf = k & 1;
        if (k + 1 < NK) load_tile((k + 1) * BK);   // overlaps WMMAs below

        // ---- A fragments for this wave (rows wave*16 .. +15) --------------
        FragU a_hi, a_lo;
        {
            const int base = (wave * 16 + l16) * LDSTR;
            a_hi.h[0] = *reinterpret_cast<const v8bf*>(&sa_hi[buf][base + hs * 8]);
            a_hi.h[1] = *reinterpret_cast<const v8bf*>(&sa_hi[buf][base + 16 + hs * 8]);
            a_lo.h[0] = *reinterpret_cast<const v8bf*>(&sa_lo[buf][base + hs * 8]);
            a_lo.h[1] = *reinterpret_cast<const v8bf*>(&sa_lo[buf][base + 16 + hs * 8]);
        }
        for (int j = 0; j < 4; ++j) {
            const int base = (j * 16 + l16) * LDSTR;
            FragU b_hi, b_lo;
            b_hi.h[0] = *reinterpret_cast<const v8bf*>(&sb_hi[buf][base + hs * 8]);
            b_hi.h[1] = *reinterpret_cast<const v8bf*>(&sb_hi[buf][base + 16 + hs * 8]);
            b_lo.h[0] = *reinterpret_cast<const v8bf*>(&sb_lo[buf][base + hs * 8]);
            b_lo.h[1] = *reinterpret_cast<const v8bf*>(&sb_lo[buf][base + 16 + hs * 8]);
            // acc += hi*hi + hi*lo + lo*hi  (lo*lo ~ 2^-16, negligible)
            acc[j] = __builtin_amdgcn_wmma_f32_16x16x32_bf16(false, a_hi.v, false, b_hi.v, (short)0, acc[j], false, false);
            acc[j] = __builtin_amdgcn_wmma_f32_16x16x32_bf16(false, a_hi.v, false, b_lo.v, (short)0, acc[j], false, false);
            acc[j] = __builtin_amdgcn_wmma_f32_16x16x32_bf16(false, a_lo.v, false, b_hi.v, (short)0, acc[j], false, false);
        }

        if (k + 1 < NK) {
            split_store(buf ^ 1);   // other buffer: no wave still reads it
            __syncthreads();        // single barrier per K-step
        }
    }

    // ---- epilogue: bias + relu, fp32 store -------------------------------
    // C/D layout: elem r -> (M = hs*8 + r, N = l16) within the 16x16 tile.
    for (int j = 0; j < 4; ++j) {
        const int col = n0 + j * 16 + l16;
        const float bv = bias[col];
        for (int r = 0; r < 8; ++r) {
            const int row = m0 + wave * 16 + hs * 8 + r;
            H[(size_t)row * DIM + col] = fmaxf(acc[j][r] + bv, 0.0f);
        }
    }
}

// cnt_subj[i] = #{r: subj==i}, cnt_obj[i] = #{r: obj==i}
__global__ __launch_bounds__(256)
void rpcm_counts(const int* __restrict__ rel, float* __restrict__ cnt_s,
                 float* __restrict__ cnt_o)
{
    __shared__ int cs[N_OBJ], co[N_OBJ];
    const int t = threadIdx.x;
    cs[t] = 0; co[t] = 0;
    __syncthreads();
    for (int r = t; r < N_REL; r += 256) {
        atomicAdd(&cs[rel[2 * r + 0]], 1);
        atomicAdd(&co[rel[2 * r + 1]], 1);
    }
    __syncthreads();
    cnt_s[t] = (float)cs[t];
    cnt_o[t] = (float)co[t];
}

// deg[r] = #{j != r : relations share an endpoint}   (pred_pred rowsum)
__global__ __launch_bounds__(256)
void rpcm_deg(const int* __restrict__ rel, float* __restrict__ deg)
{
    __shared__ int s0[N_REL], s1[N_REL];
    const int t = threadIdx.x;
    const int r = blockIdx.x * 256 + t;
    for (int i = t; i < N_REL; i += 256) { s0[i] = rel[2 * i]; s1[i] = rel[2 * i + 1]; }
    __syncthreads();
    const int a0 = s0[r], a1 = s1[r];
    int d = 0;
    for (int j = 0; j < N_REL; ++j) {
        const bool m = (a0 == s0[j]) | (a1 == s1[j]) | (a0 == s1[j]) | (a1 == s0[j]);
        d += (m && (j != r)) ? 1 : 0;
    }
    deg[r] = (float)d;
}

// colsum[d] = sum_i h4[i][d]  (obj_obj = 1-I  =>  attn@h = colsum - self)
__global__ __launch_bounds__(256)
void rpcm_colsum(const float* __restrict__ h4, float* __restrict__ colsum)
{
    const int d = blockIdx.x * 256 + threadIdx.x;
    float s = 0.f;
    for (int i = 0; i < N_OBJ; ++i) s += h4[(size_t)i * DIM + d];
    colsum[d] = s;
}

__global__ __launch_bounds__(256)
void rpcm_obj_out(const float* __restrict__ x_obj, const int* __restrict__ rel,
                  const float* __restrict__ h0, const float* __restrict__ h1,
                  const float* __restrict__ h4, const float* __restrict__ colsum,
                  const float* __restrict__ cnt_s, const float* __restrict__ cnt_o,
                  float* __restrict__ out)
{
    __shared__ int lsub[N_REL], lobj[N_REL];
    __shared__ int nsub, nobj;
    const int i = blockIdx.x;
    const int t = threadIdx.x;
    if (t == 0) {  // deterministic ascending neighbor lists
        int ns = 0, no = 0;
        for (int r = 0; r < N_REL; ++r) {
            if (rel[2 * r]     == i) lsub[ns++] = r;
            if (rel[2 * r + 1] == i) lobj[no++] = r;
        }
        nsub = ns; nobj = no;
    }
    __syncthreads();
    const float inv_oo = 1.f / (255.f + EPSF);
    const float inv_s  = 1.f / (cnt_s[i] + EPSF);
    const float inv_o  = 1.f / (cnt_o[i] + EPSF);
    for (int c = 0; c < DIM / 256; ++c) {
        const int d = c * 256 + t;
        float as = 0.f, ao = 0.f;
        for (int q = 0; q < nsub; ++q) as += h0[(size_t)lsub[q] * DIM + d];
        for (int q = 0; q < nobj; ++q) ao += h1[(size_t)lobj[q] * DIM + d];
        const float so = (colsum[d] - h4[(size_t)i * DIM + d]) * inv_oo;
        out[(size_t)i * DIM + d] = x_obj[(size_t)i * DIM + d]
                                 + (so + as * inv_s + ao * inv_o) * (1.f / 3.f);
    }
}

__global__ __launch_bounds__(256)
void rpcm_pred_out(const float* __restrict__ x_pred, const int* __restrict__ rel,
                   const float* __restrict__ h2, const float* __restrict__ h3,
                   const float* __restrict__ h5, const float* __restrict__ deg,
                   float* __restrict__ out)
{
    __shared__ int match[N_REL];
    __shared__ int nm;
    const int r = blockIdx.x;
    const int t = threadIdx.x;
    if (t == 0) {  // deterministic ascending match list for pred_pred row r
        const int a0 = rel[2 * r], a1 = rel[2 * r + 1];
        int n = 0;
        for (int j = 0; j < N_REL; ++j) {
            const int b0 = rel[2 * j], b1 = rel[2 * j + 1];
            const bool m = (a0 == b0) | (a1 == b1) | (a0 == b1) | (a1 == b0);
            if (m && (j != r)) match[n++] = j;
        }
        nm = n;
    }
    __syncthreads();
    const int   si   = rel[2 * r], oi = rel[2 * r + 1];
    const float inv1 = 1.f / (1.f + EPSF);      // subj_pred^T / obj_pred^T rowsum == 1
    const float invd = 1.f / (deg[r] + EPSF);
    for (int c = 0; c < DIM / 256; ++c) {
        const int d = c * 256 + t;
        float ar = 0.f;
        for (int q = 0; q < nm; ++q) ar += h5[(size_t)match[q] * DIM + d];
        out[(size_t)r * DIM + d] = x_pred[(size_t)r * DIM + d]
            + (h2[(size_t)si * DIM + d] * inv1 + h3[(size_t)oi * DIM + d] * inv1
               + ar * invd) * (1.f / 3.f);
    }
}

extern "C" void kernel_launch(void* const* d_in, const int* in_sizes, int n_in,
                              void* d_out, int out_size, void* d_ws, size_t ws_size,
                              hipStream_t stream)
{
    (void)in_sizes; (void)n_in; (void)out_size; (void)ws_size;

    const float* obj_feats  = (const float*)d_in[0];   // 256 x 4096
    const float* pred_feats = (const float*)d_in[1];   // 1024 x 4096
    const int*   rel_inds   = (const int*)  d_in[2];   // 1024 x 2
    const float* Wc         = (const float*)d_in[3];   // 6 x 4096 x 4096
    const float* bc         = (const float*)d_in[4];   // 6 x 4096

    float* out_obj  = (float*)d_out;
    float* out_pred = out_obj + (size_t)N_OBJ * DIM;

    // workspace layout (~63 MB fp32, fully written before any read)
    float* ws  = (float*)d_ws;
    float* h0  = ws;                          // 1024 x 4096
    float* h1  = h0 + (size_t)N_REL * DIM;
    float* h5  = h1 + (size_t)N_REL * DIM;
    float* h2  = h5 + (size_t)N_REL * DIM;    // 256 x 4096
    float* h3  = h2 + (size_t)N_OBJ * DIM;
    float* h4  = h3 + (size_t)N_OBJ * DIM;
    float* csm = h4 + (size_t)N_OBJ * DIM;    // 4096
    float* cs  = csm + DIM;                   // 256
    float* co  = cs + N_OBJ;                  // 256
    float* dg  = co + N_OBJ;                  // 1024

    const size_t WSZ = (size_t)DIM * DIM;
    const dim3 blk(128);
    const dim3 gP(DIM / BN, N_REL / BM);      // 64 x 16
    const dim3 gO(DIM / BN, N_OBJ / BM);      // 64 x 4

    rpcm_gemm_relu<<<gP, blk, 0, stream>>>(pred_feats, Wc + 0 * WSZ, bc + 0 * DIM, h0);
    rpcm_gemm_relu<<<gP, blk, 0, stream>>>(pred_feats, Wc + 1 * WSZ, bc + 1 * DIM, h1);
    rpcm_gemm_relu<<<gO, blk, 0, stream>>>(obj_feats,  Wc + 2 * WSZ, bc + 2 * DIM, h2);
    rpcm_gemm_relu<<<gO, blk, 0, stream>>>(obj_feats,  Wc + 3 * WSZ, bc + 3 * DIM, h3);
    rpcm_gemm_relu<<<gO, blk, 0, stream>>>(obj_feats,  Wc + 4 * WSZ, bc + 4 * DIM, h4);
    rpcm_gemm_relu<<<gP, blk, 0, stream>>>(pred_feats, Wc + 5 * WSZ, bc + 5 * DIM, h5);

    rpcm_counts<<<1, 256, 0, stream>>>(rel_inds, cs, co);
    rpcm_deg<<<4, 256, 0, stream>>>(rel_inds, dg);
    rpcm_colsum<<<DIM / 256, 256, 0, stream>>>(h4, csm);
    rpcm_obj_out<<<N_OBJ, 256, 0, stream>>>(obj_feats, rel_inds, h0, h1, h4, csm, cs, co, out_obj);
    rpcm_pred_out<<<N_REL, 256, 0, stream>>>(pred_feats, rel_inds, h2, h3, h5, dg, out_pred);
}
